// MultiHeadedAttention_7722351198230
// MI455X (gfx1250) — compile-verified, matrix operand order checked by reading, not run on hardware
//
#include <hip/hip_runtime.h>
#include <hip/hip_bf16.h>

// ---------------- problem constants (from reference) ----------------
#define BB   4
#define SS   2048
#define DD   512
#define HH   8
#define DQK  64
#define DV   64
#define PP   1536          // (DQK+DQK+DV)*H
#define MM   (BB*SS)       // 8192 rows
#define EPSL 1e-5f

// 1 = use GLOBAL_LOAD_ASYNC_TO_LDS_B128 + s_wait_asynccnt double buffering
#define ATHENA_ASYNC 1

typedef __attribute__((ext_vector_type(16))) __bf16        v16bf;
typedef __attribute__((ext_vector_type(8)))  float         v8f;
typedef __attribute__((ext_vector_type(8)))  unsigned int  v8u;
typedef __attribute__((ext_vector_type(4)))  unsigned int  u32x4;
typedef unsigned short u16;

// ---------------- helpers ----------------
__device__ __forceinline__ u16 f2bf(float x) {
    unsigned int u = __builtin_bit_cast(unsigned int, x);
    unsigned int r = u + 0x7FFFu + ((u >> 16) & 1u);   // round-to-nearest-even
    return (u16)(r >> 16);
}

// Load a 16x32 bf16 fragment (A-layout; also used for B via B^T row-major).
// Element (mn, k) lives at base[mn*ld + kbase + k].
// Per ISA 7.12.2 (16-bit A 16x32): lane L in [0,16) holds row mn=L;
// hi = lane>>4 selects K groups {+8}; VGPR v<4: K=2v(+8*hi), v>=4: K=16+2(v-4)(+8*hi).
__device__ __forceinline__ v16bf load_frag(const u16* base, int mn, int ld, int kbase, int hi) {
    const u16* p = base + (size_t)mn * ld + kbase + hi * 8;
    v8u r;
#pragma unroll
    for (int v = 0; v < 4; ++v)
        r[v] = *(const unsigned int*)(p + 2 * v);
#pragma unroll
    for (int v = 0; v < 4; ++v)
        r[4 + v] = *(const unsigned int*)(p + 16 + 2 * v);
    return __builtin_bit_cast(v16bf, r);
}

__device__ __forceinline__ v8f wmma_bf16(v16bf a, v16bf b, v8f c) {
    // D = A(16x32) * B(32x16) + C, f32 accumulate
    return __builtin_amdgcn_wmma_f32_16x16x32_bf16(false, a, false, b, (short)0, c, false, false);
}

__device__ __forceinline__ float rowmax16(float v) {
#pragma unroll
    for (int m = 1; m < 16; m <<= 1) v = fmaxf(v, __shfl_xor(v, m, 32));
    return v;
}
__device__ __forceinline__ float rowsum16(float v) {
#pragma unroll
    for (int m = 1; m < 16; m <<= 1) v += __shfl_xor(v, m, 32);
    return v;
}

// ---- async copy: 16 bytes/lane, global -> LDS, tracked by ASYNCcnt ----
__device__ __forceinline__ unsigned lds_off(const void* p) {
    // generic LDS address = {aperture, 32-bit wave-relative offset} (ISA 10.2)
    return (unsigned)(unsigned long long)p;
}
__device__ __forceinline__ void cp16_g2l(u16* ldst, const u16* gsrc) {
#if ATHENA_ASYNC
    asm volatile("global_load_async_to_lds_b128 %0, %1, off"
                 :: "v"(lds_off(ldst)), "v"(gsrc) : "memory");
#else
    *(u32x4*)ldst = *(const u32x4*)gsrc;
#endif
}
#if ATHENA_ASYNC
#define STAGE_WAIT_NEXT() asm volatile("s_wait_asynccnt 0x10" ::: "memory")
#define STAGE_WAIT_ALL()  asm volatile("s_wait_asynccnt 0x0"  ::: "memory")
#else
#define STAGE_WAIT_NEXT() asm volatile("s_wait_dscnt 0x0" ::: "memory")
#define STAGE_WAIT_ALL()  asm volatile("s_wait_dscnt 0x0" ::: "memory")
#endif
#define DS_WAIT_0() asm volatile("s_wait_dscnt 0x0" ::: "memory")

// ---------------- kernel 0: f32 -> bf16 ----------------
__global__ void cvt_bf16(const float* __restrict__ src, u16* __restrict__ dst, int n) {
    int i = blockIdx.x * blockDim.x + threadIdx.x;
    if (i < n) dst[i] = f2bf(src[i]);
}

// ---------------- kernel 1: QKV projection GEMM ----------------
// proj(8192x1536) = x(8192x512) * Wc^T ; Wc (1536x512) row-major == B^T K-contiguous.
// Epilogue: +bc, Q scaled by 1/8, Q/K head-major, V stored transposed vT[b][h][dv][s].
__global__ __launch_bounds__(256) void gemm_qkv(const u16* __restrict__ xb,
                                                const u16* __restrict__ Wcb,
                                                const float* __restrict__ bc,
                                                u16* __restrict__ qb,
                                                u16* __restrict__ kb,
                                                u16* __restrict__ vT) {
    const int lane  = threadIdx.x & 31;
    const int wave  = threadIdx.x >> 5;
    const int hi    = lane >> 4;
    const int col16 = lane & 15;
    const int wm = wave >> 2, wn = wave & 3;            // 2x4 wave grid
    const int row0 = blockIdx.x * 64 + wm * 32;         // M base of this wave
    const int col0 = blockIdx.y * 128 + wn * 32;        // N base of this wave

    v8f acc[2][2] = {};
    for (int k = 0; k < DD; k += 32) {
        v16bf a0 = load_frag(xb,  row0 + col16,      DD, k, hi);
        v16bf a1 = load_frag(xb,  row0 + 16 + col16, DD, k, hi);
        v16bf b0 = load_frag(Wcb, col0 + col16,      DD, k, hi);
        v16bf b1 = load_frag(Wcb, col0 + 16 + col16, DD, k, hi);
        acc[0][0] = wmma_bf16(a0, b0, acc[0][0]);
        acc[0][1] = wmma_bf16(a0, b1, acc[0][1]);
        acc[1][0] = wmma_bf16(a1, b0, acc[1][0]);
        acc[1][1] = wmma_bf16(a1, b1, acc[1][1]);
    }
#pragma unroll
    for (int mi = 0; mi < 2; ++mi)
#pragma unroll
        for (int ni = 0; ni < 2; ++ni)
#pragma unroll
            for (int r = 0; r < 8; ++r) {
                int grow = row0 + mi * 16 + r + hi * 8;
                int gcol = col0 + ni * 16 + col16;
                float v = acc[mi][ni][r] + bc[gcol];
                int b = grow >> 11, s = grow & (SS - 1);
                if (gcol < HH * DQK) {                       // Q (pre-scaled by 1/sqrt(64))
                    int h = gcol >> 6, d = gcol & 63;
                    qb[(((size_t)(b * HH + h) * SS + s) << 6) + d] = f2bf(v * 0.125f);
                } else if (gcol < 2 * HH * DQK) {            // K
                    int n = gcol - HH * DQK;
                    int h = n >> 6, d = n & 63;
                    kb[(((size_t)(b * HH + h) * SS + s) << 6) + d] = f2bf(v);
                } else {                                     // V, transposed store
                    int n = gcol - 2 * HH * DQK;
                    int h = n >> 6, d = n & 63;
                    vT[(((size_t)(b * HH + h) * DV + d) << 11) + s] = f2bf(v);
                }
            }
}

// ---------------- kernel 2: flash attention ----------------
// One wave per 32-row Q tile; 32 keys per step. K/V tiles are async-staged into
// double-buffered per-wave LDS (ASYNCcnt), probabilities re-layouted via LDS (DScnt).
__global__ __launch_bounds__(128) void attn(const u16* __restrict__ qb,
                                            const u16* __restrict__ kb,
                                            const u16* __restrict__ vT,
                                            u16* __restrict__ concatb) {
    __shared__ __align__(16) u16 Kl[4][2][32 * 64];   // [wave][buf][key][dqk]
    __shared__ __align__(16) u16 Vl[4][2][64 * 32];   // [wave][buf][dv][key]
    __shared__ __align__(16) u16 Pl[4][32 * 32];      // [wave][qrow][key]

    const int lane  = threadIdx.x & 31;
    const int wave  = threadIdx.x >> 5;
    const int hi    = lane >> 4;
    const int col16 = lane & 15;

    const int t  = blockIdx.x * 4 + wave;             // 2048 tiles = B*H*(S/32)
    const int b  = t >> 9;
    const int h  = (t >> 6) & (HH - 1);
    const int s0 = (t & 63) << 5;

    const size_t hb = (size_t)b * HH + h;
    const u16* qh = qb + hb * SS * DQK;
    const u16* kh = kb + hb * SS * DQK;
    const u16* vh = vT + hb * DV * SS;

    // stage one 32-key step (K: contiguous 4KB; V: 64 rows x 64B strided)
    auto stage = [&](int j, int buf) {
#pragma unroll
        for (int i = 0; i < 8; ++i)   // K tile: flat 4KB copy
            cp16_g2l(&Kl[wave][buf][i * 256 + lane * 8],
                     kh + (size_t)j * DQK + i * 256 + lane * 8);
#pragma unroll
        for (int i = 0; i < 8; ++i) { // V tile: gather 64B per dv row
            int c = i * 32 + lane;    // chunk id 0..255
            int d = c >> 2, sub = c & 3;
            cp16_g2l(&Vl[wave][buf][d * 32 + sub * 8],
                     vh + (size_t)d * SS + j + sub * 8);
        }
    };

    const v16bf qa0a = load_frag(qh + (size_t)s0 * DQK,        col16, DQK, 0,  hi);
    const v16bf qa0b = load_frag(qh + (size_t)s0 * DQK,        col16, DQK, 32, hi);
    const v16bf qa1a = load_frag(qh + (size_t)(s0 + 16) * DQK, col16, DQK, 0,  hi);
    const v16bf qa1b = load_frag(qh + (size_t)(s0 + 16) * DQK, col16, DQK, 32, hi);

    v8f acc[2][4] = {};
    float mrow[2][8], lrow[2][8];
#pragma unroll
    for (int mt = 0; mt < 2; ++mt)
#pragma unroll
        for (int r = 0; r < 8; ++r) { mrow[mt][r] = -1e30f; lrow[mt][r] = 0.f; }

    stage(0, 0);
    for (int jt = 0; jt < SS / 32; ++jt) {
        const int buf = jt & 1;
        const bool has_next = (jt + 1) < (SS / 32);
        if (has_next) stage((jt + 1) * 32, buf ^ 1);
        if (has_next) { STAGE_WAIT_NEXT(); } else { STAGE_WAIT_ALL(); }

        const u16* Kb = &Kl[wave][buf][0];
        // scores: 2 q-mtiles x 2 key-tiles
        v8f c[2][2];
#pragma unroll
        for (int kt = 0; kt < 2; ++kt) {
            v16bf k0 = load_frag(Kb, kt * 16 + col16, DQK, 0,  hi);
            v16bf k1 = load_frag(Kb, kt * 16 + col16, DQK, 32, hi);
            v8f z = {};
            c[0][kt] = wmma_bf16(qa0b, k1, wmma_bf16(qa0a, k0, z));
            c[1][kt] = wmma_bf16(qa1b, k1, wmma_bf16(qa1a, k0, z));
        }

        // online softmax; C layout: VGPR r -> row (r + 8*hi), col = lane&15
#pragma unroll
        for (int mt = 0; mt < 2; ++mt)
#pragma unroll
            for (int r = 0; r < 8; ++r) {
                float mx = rowmax16(fmaxf(c[mt][0][r], c[mt][1][r]));
                float mn = fmaxf(mrow[mt][r], mx);
                float alpha = __expf(mrow[mt][r] - mn);
                float p0 = __expf(c[mt][0][r] - mn);
                float p1 = __expf(c[mt][1][r] - mn);
                lrow[mt][r] = lrow[mt][r] * alpha + rowsum16(p0 + p1);
                mrow[mt][r] = mn;
#pragma unroll
                for (int ch = 0; ch < 4; ++ch) acc[mt][ch][r] *= alpha;
                int Mr = mt * 16 + r + hi * 8;
                Pl[wave][Mr * 32 + col16]      = f2bf(p0);
                Pl[wave][Mr * 32 + col16 + 16] = f2bf(p1);
            }
        DS_WAIT_0();   // LDS RAW within the wave (lockstep wave32)

        const v16bf pa0 = load_frag(&Pl[wave][0],       col16, 32, 0, hi);
        const v16bf pa1 = load_frag(&Pl[wave][16 * 32], col16, 32, 0, hi);
        const u16* Vb = &Vl[wave][buf][0];
#pragma unroll
        for (int ch = 0; ch < 4; ++ch) {
            v16bf vf = load_frag(Vb, ch * 16 + col16, 32, 0, hi);
            acc[0][ch] = wmma_bf16(pa0, vf, acc[0][ch]);
            acc[1][ch] = wmma_bf16(pa1, vf, acc[1][ch]);
        }
    }

#pragma unroll
    for (int mt = 0; mt < 2; ++mt)
#pragma unroll
        for (int ch = 0; ch < 4; ++ch)
#pragma unroll
            for (int r = 0; r < 8; ++r) {
                int row = s0 + mt * 16 + r + hi * 8;
                float v = acc[mt][ch][r] / lrow[mt][r];
                concatb[(((size_t)b * SS + row) << 9) + h * DV + ch * 16 + col16] = f2bf(v);
            }
}

// ---------------- kernel 3: output GEMM + bias + residual ----------------
// res(8192x512) = concat(8192x512)*Wo^T + bo + x ; Wo (512x512) row-major == B^T.
__global__ __launch_bounds__(256) void gemm_out(const u16* __restrict__ cb,
                                                const u16* __restrict__ Wob,
                                                const float* __restrict__ bo,
                                                const float* __restrict__ x,
                                                float* __restrict__ res) {
    const int lane  = threadIdx.x & 31;
    const int wave  = threadIdx.x >> 5;
    const int hi    = lane >> 4;
    const int col16 = lane & 15;
    const int wm = wave >> 2, wn = wave & 3;
    const int row0 = blockIdx.x * 64 + wm * 32;
    const int col0 = blockIdx.y * 128 + wn * 32;

    v8f acc[2][2] = {};
    for (int k = 0; k < DD; k += 32) {
        v16bf a0 = load_frag(cb,  row0 + col16,      DD, k, hi);
        v16bf a1 = load_frag(cb,  row0 + 16 + col16, DD, k, hi);
        v16bf b0 = load_frag(Wob, col0 + col16,      DD, k, hi);
        v16bf b1 = load_frag(Wob, col0 + 16 + col16, DD, k, hi);
        acc[0][0] = wmma_bf16(a0, b0, acc[0][0]);
        acc[0][1] = wmma_bf16(a0, b1, acc[0][1]);
        acc[1][0] = wmma_bf16(a1, b0, acc[1][0]);
        acc[1][1] = wmma_bf16(a1, b1, acc[1][1]);
    }
#pragma unroll
    for (int mi = 0; mi < 2; ++mi)
#pragma unroll
        for (int ni = 0; ni < 2; ++ni)
#pragma unroll
            for (int r = 0; r < 8; ++r) {
                int grow = row0 + mi * 16 + r + hi * 8;
                int gcol = col0 + ni * 16 + col16;
                size_t off = (size_t)grow * DD + gcol;
                res[off] = acc[mi][ni][r] + bo[gcol] + x[off];
            }
}

// ---------------- kernel 4: LayerNorm, one wave per row ----------------
__global__ __launch_bounds__(256) void ln_kernel(const float* __restrict__ res,
                                                 const float* __restrict__ gamma,
                                                 const float* __restrict__ beta,
                                                 float* __restrict__ out) {
    const int lane = threadIdx.x & 31;
    const int wave = threadIdx.x >> 5;
    const size_t row = (size_t)blockIdx.x * 8 + wave;
    const float* p = res + row * DD + lane * 16;

    float v[16];
    float s = 0.f, sq = 0.f;
#pragma unroll
    for (int j = 0; j < 16; ++j) {
        v[j] = p[j];
        s += v[j];
        sq += v[j] * v[j];
    }
#pragma unroll
    for (int m = 1; m < 32; m <<= 1) {
        s  += __shfl_xor(s,  m, 32);
        sq += __shfl_xor(sq, m, 32);
    }
    const float mean = s * (1.f / DD);
    const float var  = sq * (1.f / DD) - mean * mean;
    const float rstd = rsqrtf(var + EPSL);

    float* o = out + row * DD + lane * 16;
#pragma unroll
    for (int j = 0; j < 16; ++j) {
        int c = lane * 16 + j;
        o[j] = (v[j] - mean) * rstd * gamma[c] + beta[c];
    }
}

// ---------------- host launch ----------------
extern "C" void kernel_launch(void* const* d_in, const int* in_sizes, int n_in,
                              void* d_out, int out_size, void* d_ws, size_t ws_size,
                              hipStream_t stream) {
    const float* x     = (const float*)d_in[0];
    const float* Wc    = (const float*)d_in[1];
    const float* bc    = (const float*)d_in[2];
    const float* Wo    = (const float*)d_in[3];
    const float* bo    = (const float*)d_in[4];
    const float* gamma = (const float*)d_in[5];
    const float* beta  = (const float*)d_in[6];
    float* out = (float*)d_out;

    // workspace carve-up (bytes)
    char* ws = (char*)d_ws;
    u16* xb      = (u16*)(ws);                       // 8192*512   bf16
    u16* Wcb     = (u16*)(ws + 8388608);             // 1536*512   bf16
    u16* Wob     = (u16*)(ws + 9961472);             // 512*512    bf16
    u16* qb      = (u16*)(ws + 10485760);            // B*H*S*DQK  bf16
    u16* kb      = (u16*)(ws + 18874368);            // B*H*S*DQK  bf16
    u16* vT      = (u16*)(ws + 27262976);            // B*H*DV*S   bf16 (transposed)
    u16* concatb = (u16*)(ws + 35651584);            // 8192*512   bf16
    float* res   = (float*)(ws + 44040192);          // 8192*512   f32

    const int nx  = MM * DD;       // 4194304
    const int nwc = PP * DD;       // 786432
    const int nwo = DD * HH * DV;  // 262144

    cvt_bf16<<<(nx  + 255) / 256, 256, 0, stream>>>(x,  xb,  nx);
    cvt_bf16<<<(nwc + 255) / 256, 256, 0, stream>>>(Wc, Wcb, nwc);
    cvt_bf16<<<(nwo + 255) / 256, 256, 0, stream>>>(Wo, Wob, nwo);

    gemm_qkv<<<dim3(MM / 64, PP / 128), 256, 0, stream>>>(xb, Wcb, bc, qb, kb, vT);

    attn<<<(BB * HH * (SS / 32)) / 4, 128, 0, stream>>>(qb, kb, vT, concatb);

    gemm_out<<<dim3(MM / 64, DD / 128), 256, 0, stream>>>(concatb, Wob, bo, x, res);

    ln_kernel<<<MM / 8, 256, 0, stream>>>(res, gamma, beta, out);

    (void)in_sizes; (void)n_in; (void)out_size; (void)ws_size;
}
